// Self_Attention_3728031613618
// MI455X (gfx1250) — matrix operand order repaired and checked by reading, not compiled, not run on hardware
//
#include <hip/hip_runtime.h>
#include <hip/hip_bf16.h>
#include <stdint.h>

// ---------------------------------------------------------------------------
// Self-attention for MI455X (gfx1250, wave32, WMMA + TDM).
//   q,k,v = x @ W{0,1,2}            (f32 -> bf16, manual staged GEMM)
//   s     = q @ k^T / sqrt(300)     (bf16 GEMM; A/B tiles DMA'd by TDM)
//   p     = softmax(s)              (row kernel, in place)
//   out   = p @ v                   (bf16 GEMM -> f32; A tile via TDM)
// All GEMMs: v_wmma_f32_16x16x32_bf16, f32 accumulate.
// Workspace: 3*4*2048*768*2 + 4*2048*2048*2 bytes ~= 71.3 MB.
// ---------------------------------------------------------------------------

typedef __attribute__((ext_vector_type(16))) __bf16        v16bf;
typedef __attribute__((ext_vector_type(8)))  float         v8f;
typedef __attribute__((ext_vector_type(4)))  unsigned int  v4u;
typedef __attribute__((ext_vector_type(8)))  int           v8i;
typedef __attribute__((ext_vector_type(4)))  int           v4i;

#if __has_builtin(__builtin_amdgcn_tensor_load_to_lds) && \
    __has_builtin(__builtin_amdgcn_s_wait_tensorcnt)
#define ATTN_HAS_TDM 1
#else
#define ATTN_HAS_TDM 0
#endif

__device__ __forceinline__ unsigned short f32_to_bf16(float f) {
  union { float f; uint32_t u; } x; x.f = f;
  uint32_t u = x.u;
  return (unsigned short)((u + 0x7FFFu + ((u >> 16) & 1u)) >> 16);  // RNE
}
__device__ __forceinline__ float bf16_to_f32(unsigned short h) {
  union { float f; uint32_t u; } x; x.u = ((uint32_t)h) << 16;
  return x.f;
}

// Block tile: 128x128x32.  256 threads = 8 waves in a 4(M) x 2(N) grid,
// each wave computes a 32x64 sub-tile = 2x4 WMMA accumulators.
constexpr int BM = 128, BN = 128, BK = 32;
constexpr int LDSS = BK + 8;   // LDS row stride (bf16 elems): 80B rows, 16B aligned

// --- TDM helpers (declared always so discarded constexpr branches parse) ----
__device__ void tdm_load_tile_bf16(unsigned lds_byte_off,
                                   const unsigned short* gptr, int row_elems);
__device__ void tdm_wait_all();

#if ATTN_HAS_TDM
// DMA a 128-row x 32-elem bf16 tile (row stride row_elems) from global into
// LDS at lds_byte_off, padding +16B after each 64B row (LDS stride = 40 elems).
__device__ __forceinline__ void tdm_load_tile_bf16(unsigned lds_byte_off,
                                                   const unsigned short* gptr,
                                                   int row_elems) {
  const uint64_t ga = (uint64_t)(uintptr_t)gptr;
  v4u g0;
  g0.x = 1u;                                           // count=1, user desc
  g0.y = lds_byte_off;                                 // lds_addr
  g0.z = (unsigned)(ga & 0xFFFFFFFFu);                 // global_addr[31:0]
  g0.w = (unsigned)((ga >> 32) & 0x01FFFFFFu)          // global_addr[56:32]
         | (2u << 30);                                 // type = 2 ("image")
  const uint64_t s0 = (uint64_t)row_elems;             // tensor_dim0_stride
  v8i g1;
  g1[0] = (int)((1u << 16)      // data_size = 2 bytes
              | (1u << 20)      // pad_enable
              | (3u << 22)      // pad_interval: 16 DWORDs (one 64B row)
              | (3u << 25));    // pad_amount: 4 DWORDs (16B)
  g1[1] = (int)(((unsigned)row_elems & 0xFFFFu) << 16);          // td0 lo16
  g1[2] = (int)((((unsigned)row_elems >> 16) & 0xFFFFu)          // td0 hi16
              | ((unsigned)BM << 16));                           // td1 lo16
  g1[3] = (int)((unsigned)BK << 16);                 // td1 hi16=0, tile_dim0=32
  g1[4] = (int)BM;                                   // tile_dim1=128, tile_dim2=0
  g1[5] = (int)(s0 & 0xFFFFFFFFu);                   // stride0 lo32
  g1[6] = (int)((s0 >> 32) & 0xFFFFu);               // stride0 hi16, stride1 lo=0
  g1[7] = 0;                                         // stride1 hi = 0
  const v4i z4 = {0, 0, 0, 0};                       // 2D: groups 2/3 unused
  const v8i z8 = {0, 0, 0, 0, 0, 0, 0, 0};
  __builtin_amdgcn_tensor_load_to_lds(g0, g1, z4, z4, z8, 0);
}
__device__ __forceinline__ void tdm_wait_all() {
  __builtin_amdgcn_s_wait_tensorcnt(0);
}
#endif

template <bool A_F32, bool B_F32, bool B_KMAJOR, bool C_F32, bool A_TDM, bool B_TDM>
__global__ __launch_bounds__(256) void gemm_bf16_wmma(
    const void* __restrict__ Ap, const void* __restrict__ Bp,
    void* __restrict__ Cp,
    int M, int N, int K, int lda, int ldb, int ldc,
    long long strideA, long long strideB, long long strideC, float alpha) {
  (void)M; (void)N;
  __shared__ unsigned short ldsA[BM * LDSS];
  __shared__ unsigned short ldsB[BN * LDSS];

  const int tid  = threadIdx.x;
  const int lane = tid & 31;
  const int wave = tid >> 5;
  const int wm   = wave & 3;    // wave M index 0..3
  const int wn   = wave >> 2;   // wave N index 0..1
  const int half = lane >> 4;   // 0: lanes 0-15, 1: lanes 16-31
  const int l16  = lane & 15;

  const int m0 = blockIdx.x * BM;
  const int n0 = blockIdx.y * BN;
  const int bz = blockIdx.z;

  const float*          Af = (const float*)Ap          + (size_t)strideA * bz;
  const unsigned short* Ah = (const unsigned short*)Ap + (size_t)strideA * bz;
  const float*          Bf = (const float*)Bp          + (size_t)strideB * bz;
  const unsigned short* Bh = (const unsigned short*)Bp + (size_t)strideB * bz;

  // Flat->LDS offset: LDS aperture keeps the byte offset in addr[31:0].
  const unsigned ldsA_off = (unsigned)(uintptr_t)(void*)ldsA;
  const unsigned ldsB_off = (unsigned)(uintptr_t)(void*)ldsB;

  // A-tile loader mapping: 2 threads per row, 16 contiguous k each.
  const int a_row = tid >> 1;
  const int a_kh  = (tid & 1) * 16;
  // B-tile loader mapping (k-major source): 8 threads per k-row, 16 n each.
  const int b_k   = tid >> 3;          // 0..31
  const int b_n   = (tid & 7) * 16;    // 0..112

  v8f acc[2][4];
  const v8f vzero = {0.f,0.f,0.f,0.f,0.f,0.f,0.f,0.f};
#pragma unroll
  for (int i = 0; i < 2; ++i)
#pragma unroll
    for (int j = 0; j < 4; ++j) acc[i][j] = vzero;

  for (int kt = 0; kt < K; kt += BK) {
    // ---- stage A tile: LDS row-major [m][k] ----
    if constexpr (A_TDM) {
      if (wave == 0)
        tdm_load_tile_bf16(ldsA_off, Ah + (size_t)m0 * lda + kt, lda);
    } else if constexpr (A_F32) {
      const float* sp = Af + (size_t)(m0 + a_row) * lda + kt + a_kh;
      const float4* s4 = (const float4*)sp;
      union { float4 f4[4]; float f[16]; } av;
      av.f4[0] = s4[0]; av.f4[1] = s4[1]; av.f4[2] = s4[2]; av.f4[3] = s4[3];
      if (kt + BK < K) __builtin_prefetch(sp + BK, 0, 0);
      union { unsigned short s[16]; uint4 q[2]; } pk;
#pragma unroll
      for (int e = 0; e < 16; ++e) pk.s[e] = f32_to_bf16(av.f[e]);
      uint4* dst = (uint4*)&ldsA[a_row * LDSS + a_kh];
      dst[0] = pk.q[0]; dst[1] = pk.q[1];
    } else {
      const uint4* s4 = (const uint4*)(Ah + (size_t)(m0 + a_row) * lda + kt + a_kh);
      uint4* dst = (uint4*)&ldsA[a_row * LDSS + a_kh];
      dst[0] = s4[0]; dst[1] = s4[1];
    }

    // ---- stage B tile: LDS [n][k] (k contiguous) ----
    if constexpr (B_TDM) {
      if (wave == 1)
        tdm_load_tile_bf16(ldsB_off, Bh + (size_t)n0 * ldb + kt, ldb);
    } else if constexpr (B_KMAJOR) {
      // source is [k][n] row-major -> transpose on LDS store
      if constexpr (B_F32) {
        const float* sp = Bf + (size_t)(kt + b_k) * ldb + n0 + b_n;
        const float4* s4 = (const float4*)sp;
        union { float4 f4[4]; float f[16]; } bv;
        bv.f4[0] = s4[0]; bv.f4[1] = s4[1]; bv.f4[2] = s4[2]; bv.f4[3] = s4[3];
        if (kt + BK < K) __builtin_prefetch(sp + (size_t)BK * ldb, 0, 0);
#pragma unroll
        for (int e = 0; e < 16; ++e)
          ldsB[(b_n + e) * LDSS + b_k] = f32_to_bf16(bv.f[e]);
      } else {
        const unsigned short* sp = Bh + (size_t)(kt + b_k) * ldb + n0 + b_n;
        const uint4* s4 = (const uint4*)sp;
        union { uint4 q[2]; unsigned short s[16]; } bv;
        bv.q[0] = s4[0]; bv.q[1] = s4[1];
        if (kt + BK < K) __builtin_prefetch(sp + (size_t)BK * ldb, 0, 0);
#pragma unroll
        for (int e = 0; e < 16; ++e)
          ldsB[(b_n + e) * LDSS + b_k] = bv.s[e];
      }
    } else {
      // source is [n][k] row-major (bf16 only) -> straight copy
      const uint4* s4 = (const uint4*)(Bh + (size_t)(n0 + a_row) * ldb + kt + a_kh);
      uint4* dst = (uint4*)&ldsB[a_row * LDSS + a_kh];
      dst[0] = s4[0]; dst[1] = s4[1];
    }

    // Issuing waves drain TENSORcnt before signaling the barrier.
    if constexpr (A_TDM || B_TDM) {
      if ((A_TDM && wave == 0) || (B_TDM && wave == 1)) tdm_wait_all();
    }
    __syncthreads();

    // ---- fragments + WMMA ----
    union FragU { uint4 q[2]; v16bf v; };
    FragU fa[2], fb[4];
#pragma unroll
    for (int i = 0; i < 2; ++i) {
      // 16-bit A 16x32 layout: lane<16 holds k 0-7 & 16-23, lane>=16 k 8-15 & 24-31
      const unsigned short* ar = &ldsA[(wm * 32 + i * 16 + l16) * LDSS];
      fa[i].q[0] = *(const uint4*)(ar + half * 8);
      fa[i].q[1] = *(const uint4*)(ar + 16 + half * 8);
    }
#pragma unroll
    for (int j = 0; j < 4; ++j) {
      // 16-bit B 32x16 layout: lane<16 holds k 0-7 & 16-23 (col=lane), lane>=16 k 8-15 & 24-31
      const unsigned short* br = &ldsB[(wn * 64 + j * 16 + l16) * LDSS];
      fb[j].q[0] = *(const uint4*)(br + half * 8);
      fb[j].q[1] = *(const uint4*)(br + 16 + half * 8);
    }
#pragma unroll
    for (int i = 0; i < 2; ++i)
#pragma unroll
      for (int j = 0; j < 4; ++j)
        acc[i][j] = __builtin_amdgcn_wmma_f32_16x16x32_bf16(
            false, fa[i].v, false, fb[j].v, (short)0, acc[i][j], false, false);

    __syncthreads();
  }

  // ---- epilogue: C layout VGPR r -> M = r + 8*half, N = lane%16 ----
#pragma unroll
  for (int i = 0; i < 2; ++i)
#pragma unroll
    for (int j = 0; j < 4; ++j) {
      const int row = m0 + wm * 32 + i * 16 + half * 8;
      const int col = n0 + wn * 64 + j * 16 + l16;
      if constexpr (C_F32) {
        float* C = (float*)Cp + (size_t)strideC * bz;
#pragma unroll
        for (int r = 0; r < 8; ++r)
          C[(size_t)(row + r) * ldc + col] = acc[i][j][r] * alpha;
      } else {
        unsigned short* C = (unsigned short*)Cp + (size_t)strideC * bz;
#pragma unroll
        for (int r = 0; r < 8; ++r)
          C[(size_t)(row + r) * ldc + col] = f32_to_bf16(acc[i][j][r] * alpha);
      }
    }
}

// Row softmax over 2048 bf16 scores; one 256-thread block per row.
__global__ __launch_bounds__(256) void softmax_rows(unsigned short* __restrict__ P) {
  __shared__ float red[256];
  const int tid = threadIdx.x;
  unsigned short* p = P + (size_t)blockIdx.x * 2048;

  union { uint4 q; unsigned short s[8]; } u;
  u.q = ((const uint4*)p)[tid];
  float v[8];
  float m = -3.4e38f;
#pragma unroll
  for (int e = 0; e < 8; ++e) { v[e] = bf16_to_f32(u.s[e]); m = fmaxf(m, v[e]); }

  red[tid] = m; __syncthreads();
  for (int off = 128; off > 0; off >>= 1) {
    if (tid < off) red[tid] = fmaxf(red[tid], red[tid + off]);
    __syncthreads();
  }
  m = red[0]; __syncthreads();

  float s = 0.f;
#pragma unroll
  for (int e = 0; e < 8; ++e) { v[e] = __expf(v[e] - m); s += v[e]; }
  red[tid] = s; __syncthreads();
  for (int off = 128; off > 0; off >>= 1) {
    if (tid < off) red[tid] += red[tid + off];
    __syncthreads();
  }
  const float inv = 1.0f / red[0];

#pragma unroll
  for (int e = 0; e < 8; ++e) u.s[e] = f32_to_bf16(v[e] * inv);
  ((uint4*)p)[tid] = u.q;
}

extern "C" void kernel_launch(void* const* d_in, const int* in_sizes, int n_in,
                              void* d_out, int out_size, void* d_ws, size_t ws_size,
                              hipStream_t stream) {
  (void)in_sizes; (void)n_in; (void)out_size; (void)ws_size;
  const float* x = (const float*)d_in[0];   // [4,2048,768]
  const float* w = (const float*)d_in[1];   // [3,768,768]
  float* out = (float*)d_out;               // [4,2048,768]

  const int B = 4, S = 2048, D = 768;
  const size_t qkv = (size_t)B * S * D;     // elems per Q/K/V tensor

  unsigned short* Q  = (unsigned short*)d_ws;
  unsigned short* Km = Q + qkv;
  unsigned short* V  = Km + qkv;
  unsigned short* P  = V + qkv;             // [B, S, S] bf16 scores/probs

  const dim3 blk(256);
  const float inv_scale = 0.05773502691896258f;  // 1/sqrt(300)

  // 1) QKV projection: A = x f32 (M=8192,K=768), B = W_z f32 k-major, C bf16.
  gemm_bf16_wmma<true, true, true, false, false, false>
      <<<dim3((B * S) / BM, D / BN, 3), blk, 0, stream>>>(
          x, w, Q, B * S, D, D, D, D, D,
          0LL, (long long)D * D, (long long)B * S * D, 1.0f);

  // 2) scores = Q @ K^T / sqrt(300): both tiles are straight bf16 copies -> TDM.
  gemm_bf16_wmma<false, false, false, false, ATTN_HAS_TDM, ATTN_HAS_TDM>
      <<<dim3(S / BM, S / BN, B), blk, 0, stream>>>(
          Q, Km, P, S, S, D, D, D, S,
          (long long)S * D, (long long)S * D, (long long)S * S, inv_scale);

  // 3) softmax over each score row.
  softmax_rows<<<dim3(B * S), blk, 0, stream>>>(P);

  // 4) out = P @ V: A (P) straight copy -> TDM; B (V) transposed manually.
  gemm_bf16_wmma<false, false, true, true, ATTN_HAS_TDM, false>
      <<<dim3(S / BM, D / BN, B), blk, 0, stream>>>(
          P, V, out, S, D, S, S, D, D,
          (long long)S * S, (long long)S * D, (long long)S * D, 1.0f);
}
